// SSMDecoder_19155554140619
// MI455X (gfx1250) — compile-verified, matrix operand order checked by reading, not run on hardware
//
#include <hip/hip_runtime.h>

// ---------------- types ----------------
typedef __attribute__((ext_vector_type(16))) __bf16 v16bf;
typedef __attribute__((ext_vector_type(8)))  __bf16 v8bf;
typedef __attribute__((ext_vector_type(8)))  float  v8f;
typedef __attribute__((ext_vector_type(4)))  unsigned int v4u;
typedef __attribute__((ext_vector_type(8)))  int v8i;
typedef __attribute__((ext_vector_type(4)))  int v4i;

#define BSZ 8
#define LSEQ 4096
#define HDIM 512
#define PDIM 256
#define MROWS (BSZ * LSEQ)      // 32768
#define KDIM 512                // H (gemm1) or 2P (gemm2)
#define NDIM 512                // 2P (gemm1) or H (gemm2)
#define NCHUNK 16
#define TCHUNK 256

// fp32 -> bf16 round-to-nearest-even
__device__ __forceinline__ unsigned short f2bf(float f) {
    unsigned int u = __float_as_uint(f);
    unsigned int r = u + 0x7FFFu + ((u >> 16) & 1u);
    return (unsigned short)(r >> 16);
}
__device__ __forceinline__ unsigned int pack2bf(float a, float b) {
    return (unsigned int)f2bf(a) | ((unsigned int)f2bf(b) << 16);
}

// ---------------- TDM: DMA a 128x32 bf16 tile (row stride 512) into LDS ----------------
// D# layout per CDNA5 ISA. 2D tensor: groups 2/3 zero.
// This toolchain exposes the 6-arg (clang-23 / therock-10.0) builtin form:
//   (uint32x4 g0, int32x8 g1, int32x4 g2, int32x4 g3, int32x8 pad, i32 cpol)
#if __has_builtin(__builtin_amdgcn_tensor_load_to_lds)
#define HAVE_TDM 1
__device__ __forceinline__ void tdm_load_tile(const unsigned short* gptr, unsigned ldsOff) {
    unsigned long long ga = (unsigned long long)gptr;
    v4u g0;
    g0.x = 1u;                                                 // count=1 (valid user D#)
    g0.y = ldsOff;                                             // lds_addr [63:32]
    g0.z = (unsigned)(ga & 0xFFFFFFFFu);                       // global_addr low
    g0.w = (unsigned)((ga >> 32) & 0x01FFFFFFu) | (2u << 30);  // global_addr hi | type=2
    v8i g1;
    g1[0] = 0x00010000;        // data_size=1 (2 bytes), no multicast/pad/iterate
    g1[1] = (int)(512u << 16); // tensor_dim0 = 512 (bits 79:48, low half here)
    g1[2] = (int)(512u << 16); // tensor_dim0 hi=0 | tensor_dim1 = 512 (low half)
    g1[3] = (int)(32u << 16);  // tensor_dim1 hi=0 | tile_dim0 = 32
    g1[4] = 128;               // tile_dim1 = 128, tile_dim2 = 0 (2D)
    g1[5] = 512;               // tensor_dim0_stride = 512 (48b, low 32)
    g1[6] = 0;                 // stride hi | tensor_dim1_stride low
    g1[7] = 0;
    v4i gz4 = {0, 0, 0, 0};
    v8i gz8 = {0, 0, 0, 0, 0, 0, 0, 0};
    __builtin_amdgcn_tensor_load_to_lds(g0, g1, gz4, gz4, gz8, 0);
}
#else
#define HAVE_TDM 0
#endif

// ---------------- param prep ----------------
__global__ void prep_small(const float* __restrict__ Lre, const float* __restrict__ Lim,
                           const float* __restrict__ logstep,
                           float2* __restrict__ lamb, float2* __restrict__ lampow,
                           float2* __restrict__ fbuf) {
    int p = threadIdx.x;
    if (p >= PDIM) return;
    float st = __expf(logstep[p]);
    float c = Lre[p], d = Lim[p];
    float er = __expf(c * st);
    float ang = d * st;
    float lr = er * __cosf(ang);
    float li = er * __sinf(ang);
    lamb[p] = make_float2(lr, li);
    float ar = lr, ai = li;
    #pragma unroll
    for (int i = 0; i < 8; ++i) {   // lam^256
        float nr = ar * ar - ai * ai;
        float ni = 2.f * ar * ai;
        ar = nr; ai = ni;
    }
    lampow[p] = make_float2(ar, ai);
    float nr = lr - 1.f, ni = li;
    float inv = 1.f / (c * c + d * d);
    fbuf[p] = make_float2((nr * c + ni * d) * inv, (ni * c - nr * d) * inv);
}

__global__ __launch_bounds__(256) void prep_W1(const float* __restrict__ B,
                                               const float2* __restrict__ fbuf,
                                               unsigned short* __restrict__ W1t) {
    int idx = blockIdx.x * 256 + threadIdx.x;     // 512*512
    int n = idx >> 9, h = idx & 511;
    int p = n & 255;
    float2 f = fbuf[p];
    float b0 = B[((size_t)p * HDIM + h) * 2 + 0];
    float b1 = B[((size_t)p * HDIM + h) * 2 + 1];
    float v = (n < PDIM) ? (f.x * b0 - f.y * b1)
                         : (f.x * b1 + f.y * b0);
    W1t[idx] = f2bf(v);
}

__global__ __launch_bounds__(256) void prep_W2(const float* __restrict__ C,
                                               unsigned short* __restrict__ W2t) {
    int idx = blockIdx.x * 256 + threadIdx.x;     // 512*512
    int h = idx >> 9, k = idx & 511;
    int p = k & 255;
    float cre = C[((size_t)h * PDIM + p) * 2 + 0];
    float cim = C[((size_t)h * PDIM + p) * 2 + 1];
    float v = (k < PDIM) ? (2.f * cre) : (-2.f * cim);
    W2t[idx] = f2bf(v);
}

// ---------------- bf16 WMMA GEMM ----------------
__device__ __forceinline__ v16bf ldfrag(const unsigned short* base, int row, int half) {
    v16bf r;
    ((v8bf*)&r)[0] = *(const v8bf*)(base + row * 32 + half * 8);
    ((v8bf*)&r)[1] = *(const v8bf*)(base + row * 32 + 16 + half * 8);
    return r;
}

template <bool SKIP>
__global__ __launch_bounds__(256) void gemm_bf16(const float* __restrict__ A,
                                                 const unsigned short* __restrict__ Bt,
                                                 float* __restrict__ Out,
                                                 const float* __restrict__ X,
                                                 const float* __restrict__ Dvec) {
    __shared__ unsigned short smA[2][128 * 32];
    __shared__ unsigned short smB[2][128 * 32];

    const int tid   = threadIdx.x;
    const int lane  = tid & 31;
    const int wid   = tid >> 5;
    const int waveM = wid >> 1;
    const int waveN = wid & 1;
    const int blockN = blockIdx.x * 128;
    const int blockM = blockIdx.y * 128;

    const int lrow  = tid >> 1;          // 0..127 tile row
    const int lhalf = tid & 1;           // 16-col half of the 32-wide k slab

    v8f acc[2][4] = {};

    // A tile stage: fp32 -> bf16 through VGPRs (conversion fused into the copy)
    auto loadA = [&](int kt, int buf) {
        const float* ap = A + (size_t)(blockM + lrow) * KDIM + kt * 32 + lhalf * 16;
        float4 f0 = ((const float4*)ap)[0];
        float4 f1 = ((const float4*)ap)[1];
        float4 f2 = ((const float4*)ap)[2];
        float4 f3 = ((const float4*)ap)[3];
        unsigned int* da = (unsigned int*)&smA[buf][lrow * 32 + lhalf * 16];
        da[0] = pack2bf(f0.x, f0.y); da[1] = pack2bf(f0.z, f0.w);
        da[2] = pack2bf(f1.x, f1.y); da[3] = pack2bf(f1.z, f1.w);
        da[4] = pack2bf(f2.x, f2.y); da[5] = pack2bf(f2.z, f2.w);
        da[6] = pack2bf(f3.x, f3.y); da[7] = pack2bf(f3.z, f3.w);
        if (kt + 2 < 16)
            __builtin_prefetch(A + (size_t)(blockM + lrow) * KDIM + (kt + 2) * 32, 0, 1);
    };

    // B tile stage: one TDM DMA per stage (wave 0 only), no VGPR round-trip
    auto loadB = [&](int kt, int buf) {
#if HAVE_TDM
        if (wid == 0) {
            unsigned ldsOff = (unsigned)(unsigned long long)(void*)&smB[buf][0];
            tdm_load_tile(Bt + (size_t)blockN * KDIM + kt * 32, ldsOff);
        }
#else
        const uint4* bp = (const uint4*)(Bt + (size_t)(blockN + lrow) * KDIM + kt * 32 + lhalf * 16);
        uint4 b0 = bp[0], b1 = bp[1];
        uint4* db = (uint4*)&smB[buf][lrow * 32 + lhalf * 16];
        db[0] = b0; db[1] = b1;
#endif
    };

    loadA(0, 0);
    loadB(0, 0);
#if HAVE_TDM
    if (wid == 0) __builtin_amdgcn_s_wait_tensorcnt(0);
#endif
    __syncthreads();

    const int frow = lane & 15;
    const int half = lane >> 4;

    int buf = 0;
    for (int kt = 0; kt < 16; ++kt) {
        if (kt + 1 < 16) {
            loadB(kt + 1, buf ^ 1);   // async DMA overlaps with WMMA below
            loadA(kt + 1, buf ^ 1);
        }

        v16bf afrag[2], bfrag[4];
        #pragma unroll
        for (int i = 0; i < 2; ++i)
            afrag[i] = ldfrag(smA[buf], waveM * 32 + i * 16 + frow, half);
        #pragma unroll
        for (int j = 0; j < 4; ++j)
            bfrag[j] = ldfrag(smB[buf], waveN * 64 + j * 16 + frow, half);

        #pragma unroll
        for (int i = 0; i < 2; ++i)
            #pragma unroll
            for (int j = 0; j < 4; ++j)
                acc[i][j] = __builtin_amdgcn_wmma_f32_16x16x32_bf16(
                    false, afrag[i], false, bfrag[j], (short)0, acc[i][j], false, false);

#if HAVE_TDM
        if (wid == 0) __builtin_amdgcn_s_wait_tensorcnt(0);
#endif
        __syncthreads();
        buf ^= 1;
    }

    #pragma unroll
    for (int i = 0; i < 2; ++i) {
        #pragma unroll
        for (int j = 0; j < 4; ++j) {
            int mBase = blockM + waveM * 32 + i * 16 + half * 8;
            int n = blockN + waveN * 64 + j * 16 + frow;
            float dscale = SKIP ? Dvec[n] : 0.f;
            #pragma unroll
            for (int r = 0; r < 8; ++r) {
                size_t off = (size_t)(mBase + r) * NDIM + n;
                float v = acc[i][j][r];
                if (SKIP) v += dscale * X[off];
                Out[off] = v;
            }
        }
    }
}

// ---------------- chunked linear-recurrence scan ----------------
__global__ __launch_bounds__(256) void scan_pass1(float* __restrict__ Bu,
                                                  const float2* __restrict__ lamb,
                                                  float2* __restrict__ carry) {
    int tid = blockIdx.x * 256 + threadIdx.x;   // 8*16*256
    int p = tid & 255;
    int c = (tid >> 8) & 15;
    int b = tid >> 12;
    float2 lam = lamb[p];
    float sr = 0.f, si = 0.f;
    size_t row = ((size_t)b * LSEQ + (size_t)c * TCHUNK) * NDIM;
    for (int i = 0; i < TCHUNK; ++i) {
        float ur = Bu[row + p];
        float ui = Bu[row + PDIM + p];
        float nr = lam.x * sr - lam.y * si + ur;
        float ni = lam.x * si + lam.y * sr + ui;
        sr = nr; si = ni;
        Bu[row + p] = sr;
        Bu[row + PDIM + p] = si;
        row += NDIM;
    }
    carry[(size_t)((b * NCHUNK + c) << 8) + p] = make_float2(sr, si);
}

__global__ __launch_bounds__(256) void scan_pass2(const float2* __restrict__ lampow,
                                                  float2* __restrict__ carry) {
    int tid = blockIdx.x * 256 + threadIdx.x;   // 8*256
    int p = tid & 255;
    int b = tid >> 8;
    float2 lp = lampow[p];
    float sr = 0.f, si = 0.f;
    for (int c = 0; c < NCHUNK; ++c) {
        size_t idx = (size_t)((b * NCHUNK + c) << 8) + p;
        float2 t = carry[idx];
        carry[idx] = make_float2(sr, si);           // exclusive prefix
        float nr = lp.x * sr - lp.y * si + t.x;
        float ni = lp.x * si + lp.y * sr + t.y;
        sr = nr; si = ni;
    }
}

__global__ __launch_bounds__(256) void scan_pass3(float* __restrict__ Bu,
                                                  const float2* __restrict__ lamb,
                                                  const float2* __restrict__ carry) {
    int tid = blockIdx.x * 256 + threadIdx.x;
    int p = tid & 255;
    int c = (tid >> 8) & 15;
    int b = tid >> 12;
    if (c == 0) return;
    float2 s0 = carry[(size_t)((b * NCHUNK + c) << 8) + p];
    float2 lam = lamb[p];
    float pr = lam.x, pi = lam.y;
    size_t row = ((size_t)b * LSEQ + (size_t)c * TCHUNK) * NDIM;
    for (int i = 0; i < TCHUNK; ++i) {
        float addr_ = pr * s0.x - pi * s0.y;
        float addi  = pr * s0.y + pi * s0.x;
        Bu[row + p]        += addr_;
        Bu[row + PDIM + p] += addi;
        float npr = pr * lam.x - pi * lam.y;
        float npi = pr * lam.y + pi * lam.x;
        pr = npr; pi = npi;
        row += NDIM;
    }
}

// ---------------- launch ----------------
extern "C" void kernel_launch(void* const* d_in, const int* in_sizes, int n_in,
                              void* d_out, int out_size, void* d_ws, size_t ws_size,
                              hipStream_t stream) {
    const float* x       = (const float*)d_in[0];
    const float* Lre     = (const float*)d_in[1];
    const float* Lim     = (const float*)d_in[2];
    const float* B       = (const float*)d_in[3];
    const float* C       = (const float*)d_in[4];
    const float* D       = (const float*)d_in[5];
    const float* logstep = (const float*)d_in[6];
    float* out = (float*)d_out;

    char* ws = (char*)d_ws;
    size_t offBu    = 0;                                        // 64 MB
    size_t offW1    = offBu + (size_t)MROWS * NDIM * 4;
    size_t offW2    = offW1 + (size_t)NDIM * KDIM * 2;          // 512 KB
    size_t offLam   = offW2 + (size_t)NDIM * KDIM * 2;          // 512 KB
    size_t offPow   = offLam + PDIM * sizeof(float2);
    size_t offF     = offPow + PDIM * sizeof(float2);
    size_t offCarry = offF + PDIM * sizeof(float2);             // 256 KB

    float*          Bu     = (float*)(ws + offBu);
    unsigned short* W1t    = (unsigned short*)(ws + offW1);
    unsigned short* W2t    = (unsigned short*)(ws + offW2);
    float2*         lamb   = (float2*)(ws + offLam);
    float2*         lampow = (float2*)(ws + offPow);
    float2*         fbuf   = (float2*)(ws + offF);
    float2*         carry  = (float2*)(ws + offCarry);

    prep_small<<<1, 256, 0, stream>>>(Lre, Lim, logstep, lamb, lampow, fbuf);
    prep_W1<<<(NDIM * KDIM) / 256, 256, 0, stream>>>(B, fbuf, W1t);
    prep_W2<<<(NDIM * KDIM) / 256, 256, 0, stream>>>(C, W2t);

    dim3 gGrid(NDIM / 128, MROWS / 128);
    gemm_bf16<false><<<gGrid, 256, 0, stream>>>(x, W1t, Bu, nullptr, nullptr);

    scan_pass1<<<(BSZ * NCHUNK * PDIM) / 256, 256, 0, stream>>>(Bu, lamb, carry);
    scan_pass2<<<(BSZ * PDIM) / 256, 256, 0, stream>>>(lampow, carry);
    scan_pass3<<<(BSZ * NCHUNK * PDIM) / 256, 256, 0, stream>>>(Bu, lamb, carry);

    gemm_bf16<true><<<gGrid, 256, 0, stream>>>(Bu, W2t, out, x, D);
}
